// FieldFormer_6408091206332
// MI455X (gfx1250) — matrix operand order repaired and checked by reading, not compiled, not run on hardware
//
#include <hip/hip_runtime.h>
#include <hip/hip_bf16.h>
#include <math.h>

typedef _Float16 h16;
typedef __attribute__((ext_vector_type(2)))  _Float16 h2;
typedef __attribute__((ext_vector_type(16))) _Float16 v16h;
typedef __attribute__((ext_vector_type(8)))  float    v8f;

#define WAVES_PER_BLOCK 2
#define KTOK 32
#define DM 64
#define DF 128
#define NLAYER 2
#define NHEAD 4

// f16 weight workspace layout (offsets in halves)
#define OFF_WQKV 0        // 2*192*64 = 24576
#define OFF_WO   24576    // 2*64*64  = 8192
#define OFF_W1   32768    // 2*128*64 = 16384
#define OFF_W2   49152    // 2*64*128 = 16384
#define OFF_HW1  65536    // 64*64    = 4096

__device__ __forceinline__ v8f wmma16(v16h a, v16h b, v8f c) {
  return __builtin_amdgcn_wmma_f32_16x16x32_f16(false, a, false, b, (short)0, c,
                                                false, false);
}

// A-fragment (16x32 f16, row-major source, this lane's row fixed).
// p0 = rowbase + kbase where kbase = (lane>=16)?16:0. elems 0..7 -> K=kbase..+7,
// elems 8..15 -> K=kbase+8..+15 (zero if pad_hi, for Dh=16 padded to 32).
__device__ __forceinline__ v16h ld_a(const h16* p0, bool pad_hi) {
  v16h a;
#pragma unroll
  for (int e = 0; e < 8; ++e) a[e] = p0[e];
#pragma unroll
  for (int e = 0; e < 8; ++e) a[8 + e] = pad_hi ? (h16)0.0f : p0[8 + e];
  return a;
}
// B-fragment (32x16 f16): lane holds column n=lane&15, 16 contiguous K halves
// starting at kbase = (lane>=16)?16:0. p0 = colbase + kbase.
__device__ __forceinline__ v16h ld_b(const h16* p0) {
  v16h b;
#pragma unroll
  for (int e = 0; e < 16; ++e) b[e] = p0[e];
  return b;
}

__device__ __forceinline__ void ln_row(const float* x, h16* y,
                                       const float* g, const float* bb) {
  float mean = 0.f;
#pragma unroll 4
  for (int n = 0; n < DM; ++n) mean += x[n];
  mean *= (1.0f / DM);
  float var = 0.f;
#pragma unroll 4
  for (int n = 0; n < DM; ++n) { float d = x[n] - mean; var += d * d; }
  var *= (1.0f / DM);
  float rs = rsqrtf(var + 1e-5f);
#pragma unroll 4
  for (int n = 0; n < DM; n += 2) {
    h2 p;
    p[0] = (h16)((x[n]     - mean) * rs * g[n]     + bb[n]);
    p[1] = (h16)((x[n + 1] - mean) * rs * g[n + 1] + bb[n + 1]);
    *(h2*)&y[n] = p;  // packed ds_store_b32
  }
}

__global__ void cvt_f16_kernel(const float* __restrict__ src,
                               h16* __restrict__ dst, int n) {
  int i = blockIdx.x * 256 + threadIdx.x;
  if (i < n) dst[i] = (h16)src[i];
}

__global__ __launch_bounds__(WAVES_PER_BLOCK * 32)
void fieldformer_kernel(
    const int* __restrict__ q_lin_idx, const int* __restrict__ offsets_ijk,
    const float* __restrict__ vals, const float* __restrict__ grid_scales,
    const int* __restrict__ pNx, const int* __restrict__ pNy,
    const int* __restrict__ pNt, const float* __restrict__ log_gammas,
    const float* __restrict__ Wp, const float* __restrict__ bp,
    const float* __restrict__ bqkv, const float* __restrict__ bo,
    const float* __restrict__ b1, const float* __restrict__ b2,
    const float* __restrict__ ln1_g, const float* __restrict__ ln1_b,
    const float* __restrict__ ln2_g, const float* __restrict__ ln2_b,
    const float* __restrict__ hg, const float* __restrict__ hb,
    const float* __restrict__ hb1, const float* __restrict__ hW2,
    const float* __restrict__ hb2, const h16* __restrict__ wf16,
    float* __restrict__ out, int Btot)
{
  __shared__ h16   sh_h  [WAVES_PER_BLOCK][KTOK * DM];       // 32x64  f16
  __shared__ h16   sh_qkv[WAVES_PER_BLOCK][KTOK * 3 * DM];   // 32x192 f16 (also o / ff)
  __shared__ float sh_scr[WAVES_PER_BLOCK][KTOK * DM];       // 32x64  f32 (scores / pre-LN / z)
  __shared__ h16   sh_att[WAVES_PER_BLOCK][KTOK * KTOK];     // 32x32  f16

  const int wave = threadIdx.x >> 5;
  const int lane = threadIdx.x & 31;
  const int q    = blockIdx.x * WAVES_PER_BLOCK + wave;
  if (q >= Btot) return;

  h16*   Hbuf = sh_h[wave];
  h16*   Qbuf = sh_qkv[wave];
  float* Sbuf = sh_scr[wave];
  h16*   Abuf = sh_att[wave];

  const int row   = lane & 15;
  const int hi    = lane >> 4;
  const int abase = hi * 16;   // A-frag K base
  const int bbase = hi * 16;   // B-frag K base

  // ---------------- gather + embed (token t = lane) ----------------
  const int Nx = pNx[0], Ny = pNy[0], Nt = pNt[0];
  const int NyNt = Ny * Nt;
  const int qi = q_lin_idx[q];
  const int i0 = qi / NyNt;
  const int r0 = qi - i0 * NyNt;
  const int j0 = r0 / Nt;
  const int k0 = r0 - j0 * Nt;
  {
    const int t  = lane;
    const int di = offsets_ijk[t * 3 + 0];
    const int dj = offsets_ijk[t * 3 + 1];
    const int dk = offsets_ijk[t * 3 + 2];
    const int I  = (i0 + di) % Nx;
    const int J  = (j0 + dj) % Ny;
    const int Kc = (k0 + dk) % Nt;
    const int nb = I * NyNt + J * Nt + Kc;
    float tok[4];
    tok[0] = (float)di * grid_scales[0] * __expf(log_gammas[0]);
    tok[1] = (float)dj * grid_scales[1] * __expf(log_gammas[1]);
    tok[2] = (float)dk * grid_scales[2] * __expf(log_gammas[2]);
    tok[3] = vals[nb];
#pragma unroll 4
    for (int n = 0; n < DM; n += 2) {
      float a0 = bp[n], a1 = bp[n + 1];
#pragma unroll
      for (int c = 0; c < 4; ++c) {
        a0 += tok[c] * Wp[n * 4 + c];
        a1 += tok[c] * Wp[(n + 1) * 4 + c];
      }
      h2 p; p[0] = (h16)a0; p[1] = (h16)a1;
      *(h2*)&Hbuf[t * DM + n] = p;
    }
  }
  asm volatile("s_wait_dscnt 0" ::: "memory");

  // ---------------- transformer layers ----------------
#pragma unroll 1
  for (int l = 0; l < NLAYER; ++l) {
    const h16* wqkv = wf16 + OFF_WQKV + l * (3 * DM * DM);
    const h16* wo   = wf16 + OFF_WO   + l * (DM * DM);
    const h16* w1   = wf16 + OFF_W1   + l * (DF * DM);
    const h16* w2   = wf16 + OFF_W2   + l * (DM * DF);
    const float* bqkv_l = bqkv + l * 3 * DM;
    const float* bo_l   = bo + l * DM;
    const float* b1_l   = b1 + l * DF;
    const float* b2_l   = b2 + l * DM;

    // warm L2/WGP$ for weights used later in this layer (global_prefetch_b8)
    __builtin_prefetch(wo + (lane << 5), 0, 3);
    __builtin_prefetch(w1 + (lane << 7), 0, 3);
    __builtin_prefetch(w2 + (lane << 7), 0, 3);

    // ---- QKV: (32x64) @ (64x192), bias folded into C, q pre-scaled ----
    {
      v16h a[2][2];
#pragma unroll
      for (int mt = 0; mt < 2; ++mt)
#pragma unroll
        for (int kt = 0; kt < 2; ++kt)
          a[mt][kt] = ld_a(Hbuf + (mt * 16 + row) * DM + kt * 32 + abase, false);
      // unroll 2: second tile's loads/WMMAs fill the first tile's D->VALU
      // hazard slots (4-coexec for f16 WMMA).
#pragma unroll 2
      for (int nt = 0; nt < 12; ++nt) {
        const int col = nt * 16 + row;
        const float bias = bqkv_l[col];
        v8f c0, c1;
#pragma unroll
        for (int r = 0; r < 8; ++r) { c0[r] = bias; c1[r] = bias; }
#pragma unroll
        for (int kt = 0; kt < 2; ++kt) {
          v16h b = ld_b(wqkv + col * DM + kt * 32 + bbase);
          c0 = wmma16(a[0][kt], b, c0);
          c1 = wmma16(a[1][kt], b, c1);
        }
        // fold 1/sqrt(Dh) into q columns (nt<4)
        const float scl = (nt < 4) ? 0.25f : 1.0f;
#pragma unroll
        for (int r = 0; r < 8; ++r) {
          Qbuf[(r + 8 * hi) * 192 + col]      = (h16)(c0[r] * scl);
          Qbuf[(16 + r + 8 * hi) * 192 + col] = (h16)(c1[r] * scl);
        }
      }
    }
    asm volatile("s_wait_dscnt 0" ::: "memory");

    // ---- attention, head by head (Dh=16 padded to K=32 for scores) ----
#pragma unroll 1
    for (int hh = 0; hh < NHEAD; ++hh) {
      const int qo = hh * 16, ko = 64 + hh * 16, vo = 128 + hh * 16;
      {
        v16h aq[2];
#pragma unroll
        for (int mt = 0; mt < 2; ++mt)
          aq[mt] = ld_a(Qbuf + (mt * 16 + row) * 192 + qo + abase, true);
#pragma unroll
        for (int nt = 0; nt < 2; ++nt) {
          v16h bk = {};
          if (hi == 0)  // lanes 16..31 cover d=16..31 which is zero padding
            bk = ld_b(Qbuf + (nt * 16 + row) * 192 + ko);
          v8f z = {};
          v8f s0 = wmma16(aq[0], bk, z);
          v8f s1 = wmma16(aq[1], bk, z);
          const int n = nt * 16 + row;
#pragma unroll
          for (int r = 0; r < 8; ++r) {
            Sbuf[(r + 8 * hi) * 32 + n]      = s0[r];
            Sbuf[(16 + r + 8 * hi) * 32 + n] = s1[r];
          }
        }
      }
      asm volatile("s_wait_dscnt 0" ::: "memory");
      {  // softmax, lane = row
        const int m = lane;
        float mx = -1e30f;
#pragma unroll 4
        for (int n = 0; n < 32; ++n) mx = fmaxf(mx, Sbuf[m * 32 + n]);
        float sum = 0.f;
#pragma unroll 4
        for (int n = 0; n < 32; ++n) {
          float e = __expf(Sbuf[m * 32 + n] - mx);
          sum += e;
          Sbuf[m * 32 + n] = e;
        }
        float rinv = 1.0f / sum;
#pragma unroll 4
        for (int n = 0; n < 32; n += 2) {
          h2 p;
          p[0] = (h16)(Sbuf[m * 32 + n]     * rinv);
          p[1] = (h16)(Sbuf[m * 32 + n + 1] * rinv);
          *(h2*)&Abuf[m * 32 + n] = p;  // packed ds_store_b32
        }
      }
      asm volatile("s_wait_dscnt 0" ::: "memory");
      {  // o_h = attn(32x32) @ v_h(32x16); overwrite q columns of this head
        v16h bv;
#pragma unroll
        for (int e = 0; e < 16; ++e)
          bv[e] = Qbuf[(bbase + e) * 192 + vo + row];
#pragma unroll
        for (int mt = 0; mt < 2; ++mt) {
          v16h aa = ld_a(Abuf + (mt * 16 + row) * 32 + abase, false);
          v8f z = {};
          v8f o = wmma16(aa, bv, z);
          const int n = qo + row;
#pragma unroll
          for (int r = 0; r < 8; ++r)
            Qbuf[(mt * 16 + r + 8 * hi) * 192 + n] = (h16)o[r];
        }
      }
      asm volatile("s_wait_dscnt 0" ::: "memory");
    }

    // ---- out-proj; bias + residual folded into C -> Sbuf (f32 pre-LN1) ----
    {
      v16h a[2][2];
#pragma unroll
      for (int mt = 0; mt < 2; ++mt)
#pragma unroll
        for (int kt = 0; kt < 2; ++kt)
          a[mt][kt] = ld_a(Qbuf + (mt * 16 + row) * 192 + kt * 32 + abase, false);
#pragma unroll 2
      for (int nt = 0; nt < 4; ++nt) {
        const int col = nt * 16 + row;
        const float bias = bo_l[col];
        v8f c0, c1;
#pragma unroll
        for (int r = 0; r < 8; ++r) {
          c0[r] = bias + (float)Hbuf[(r + 8 * hi) * DM + col];
          c1[r] = bias + (float)Hbuf[(16 + r + 8 * hi) * DM + col];
        }
#pragma unroll
        for (int kt = 0; kt < 2; ++kt) {
          v16h b = ld_b(wo + col * DM + kt * 32 + bbase);
          c0 = wmma16(a[0][kt], b, c0);
          c1 = wmma16(a[1][kt], b, c1);
        }
#pragma unroll
        for (int r = 0; r < 8; ++r) {
          Sbuf[(r + 8 * hi) * DM + col]      = c0[r];
          Sbuf[(16 + r + 8 * hi) * DM + col] = c1[r];
        }
      }
    }
    asm volatile("s_wait_dscnt 0" ::: "memory");
    ln_row(Sbuf + lane * DM, Hbuf + lane * DM, ln1_g + l * DM, ln1_b + l * DM);
    asm volatile("s_wait_dscnt 0" ::: "memory");

    // ---- FF1: relu(h @ W1^T + b1) -> Qbuf as 32x128 f16 ----
    {
      v16h a[2][2];
#pragma unroll
      for (int mt = 0; mt < 2; ++mt)
#pragma unroll
        for (int kt = 0; kt < 2; ++kt)
          a[mt][kt] = ld_a(Hbuf + (mt * 16 + row) * DM + kt * 32 + abase, false);
#pragma unroll 2
      for (int nt = 0; nt < 8; ++nt) {
        const int col = nt * 16 + row;
        const float bias = b1_l[col];
        v8f c0, c1;
#pragma unroll
        for (int r = 0; r < 8; ++r) { c0[r] = bias; c1[r] = bias; }
#pragma unroll
        for (int kt = 0; kt < 2; ++kt) {
          v16h b = ld_b(w1 + col * DM + kt * 32 + bbase);
          c0 = wmma16(a[0][kt], b, c0);
          c1 = wmma16(a[1][kt], b, c1);
        }
#pragma unroll
        for (int r = 0; r < 8; ++r) {
          Qbuf[(r + 8 * hi) * DF + col]      = (h16)fmaxf(c0[r], 0.f);
          Qbuf[(16 + r + 8 * hi) * DF + col] = (h16)fmaxf(c1[r], 0.f);
        }
      }
    }
    asm volatile("s_wait_dscnt 0" ::: "memory");

    // ---- FF2; bias + residual folded into C -> Sbuf ----
    {
      v16h a[2][4];
#pragma unroll
      for (int mt = 0; mt < 2; ++mt)
#pragma unroll
        for (int kt = 0; kt < 4; ++kt)
          a[mt][kt] = ld_a(Qbuf + (mt * 16 + row) * DF + kt * 32 + abase, false);
#pragma unroll 2
      for (int nt = 0; nt < 4; ++nt) {
        const int col = nt * 16 + row;
        const float bias = b2_l[col];
        v8f c0, c1;
#pragma unroll
        for (int r = 0; r < 8; ++r) {
          c0[r] = bias + (float)Hbuf[(r + 8 * hi) * DM + col];
          c1[r] = bias + (float)Hbuf[(16 + r + 8 * hi) * DM + col];
        }
#pragma unroll
        for (int kt = 0; kt < 4; ++kt) {
          v16h b = ld_b(w2 + col * DF + kt * 32 + bbase);
          c0 = wmma16(a[0][kt], b, c0);
          c1 = wmma16(a[1][kt], b, c1);
        }
#pragma unroll
        for (int r = 0; r < 8; ++r) {
          Sbuf[(r + 8 * hi) * DM + col]      = c0[r];
          Sbuf[(16 + r + 8 * hi) * DM + col] = c1[r];
        }
      }
    }
    asm volatile("s_wait_dscnt 0" ::: "memory");
    ln_row(Sbuf + lane * DM, Hbuf + lane * DM, ln2_g + l * DM, ln2_b + l * DM);
    asm volatile("s_wait_dscnt 0" ::: "memory");
  }

  // ---------------- head: mean over tokens, LN, MLP, dot ----------------
  {
    const int f0 = 2 * lane;
    float hm0 = 0.f, hm1 = 0.f;
#pragma unroll 4
    for (int t = 0; t < KTOK; ++t) {
      h2 hv = *(const h2*)&Hbuf[t * DM + f0];  // packed ds_load_b32
      hm0 += (float)hv[0];
      hm1 += (float)hv[1];
    }
    hm0 *= (1.0f / KTOK);
    hm1 *= (1.0f / KTOK);
    float s = hm0 + hm1;
#pragma unroll
    for (int off = 16; off >= 1; off >>= 1) s += __shfl_xor(s, off, 32);
    float mean = s * (1.0f / DM);
    float d0 = hm0 - mean, d1 = hm1 - mean;
    float v = d0 * d0 + d1 * d1;
#pragma unroll
    for (int off = 16; off >= 1; off >>= 1) v += __shfl_xor(v, off, 32);
    float rs = rsqrtf(v * (1.0f / DM) + 1e-5f);
    Sbuf[f0]     = d0 * rs * hg[f0] + hb[f0];
    Sbuf[f0 + 1] = d1 * rs * hg[f0 + 1] + hb[f0 + 1];
    asm volatile("s_wait_dscnt 0" ::: "memory");

    const h16* hw1 = wf16 + OFF_HW1;
    float po = 0.f;
#pragma unroll
    for (int gi = 0; gi < 2; ++gi) {
      const int g = f0 + gi;
      float acc = hb1[g];
#pragma unroll 4
      for (int f = 0; f < DM; ++f) acc += Sbuf[f] * (float)hw1[g * DM + f];
      float ge = 0.5f * acc * (1.0f + erff(acc * 0.70710678118654752f));
      po += ge * hW2[g];
    }
#pragma unroll
    for (int off = 16; off >= 1; off >>= 1) po += __shfl_xor(po, off, 32);
    if (lane == 0) out[q] = po + hb2[0];
  }
}

extern "C" void kernel_launch(void* const* d_in, const int* in_sizes, int n_in,
                              void* d_out, int out_size, void* d_ws, size_t ws_size,
                              hipStream_t stream) {
  h16* wf16 = (h16*)d_ws;
  // Pre-convert all GEMM weights to f16 in workspace (L2-resident thereafter).
  cvt_f16_kernel<<<(24576 + 255) / 256, 256, 0, stream>>>(
      (const float*)d_in[11], wf16 + OFF_WQKV, 24576);   // Wqkv
  cvt_f16_kernel<<<(8192 + 255) / 256, 256, 0, stream>>>(
      (const float*)d_in[13], wf16 + OFF_WO, 8192);      // Wo
  cvt_f16_kernel<<<(16384 + 255) / 256, 256, 0, stream>>>(
      (const float*)d_in[15], wf16 + OFF_W1, 16384);     // W1
  cvt_f16_kernel<<<(16384 + 255) / 256, 256, 0, stream>>>(
      (const float*)d_in[17], wf16 + OFF_W2, 16384);     // W2
  cvt_f16_kernel<<<(4096 + 255) / 256, 256, 0, stream>>>(
      (const float*)d_in[25], wf16 + OFF_HW1, 4096);     // hW1

  const int Btot = in_sizes[0];
  const int nblk = (Btot + WAVES_PER_BLOCK - 1) / WAVES_PER_BLOCK;
  fieldformer_kernel<<<nblk, WAVES_PER_BLOCK * 32, 0, stream>>>(
      (const int*)d_in[0], (const int*)d_in[1], (const float*)d_in[2],
      (const float*)d_in[4], (const int*)d_in[5], (const int*)d_in[6],
      (const int*)d_in[7], (const float*)d_in[8], (const float*)d_in[9],
      (const float*)d_in[10], (const float*)d_in[12], (const float*)d_in[14],
      (const float*)d_in[16], (const float*)d_in[18], (const float*)d_in[19],
      (const float*)d_in[20], (const float*)d_in[21], (const float*)d_in[22],
      (const float*)d_in[23], (const float*)d_in[24], (const float*)d_in[26],
      (const float*)d_in[27], (const float*)d_in[28], wf16, (float*)d_out, Btot);
}
